// GPT2Block_19885698580549
// MI455X (gfx1250) — compile-verified, matrix operand order checked by reading, not run on hardware
//
#include <hip/hip_runtime.h>
#include <hip/hip_bf16.h>
#include <math.h>

#define DIM    2048
#define NH     16
#define HD     128
#define BATCH  2
#define SEQ    2048
#define TOKENS (BATCH * SEQ)   // 4096

typedef unsigned short u16;
typedef __bf16 bf16_t;
typedef bf16_t v16bf __attribute__((ext_vector_type(16)));
typedef float  v8f   __attribute__((ext_vector_type(8)));
typedef int    v4i   __attribute__((ext_vector_type(4)));

// ---- CDNA5 async global->LDS path (guarded; falls back to reg staging) -----
#if defined(__has_builtin)
#  if __has_builtin(__builtin_amdgcn_global_load_async_to_lds_b128)
#    define USE_ASYNC_LDS 1
#  endif
#endif
#ifndef USE_ASYNC_LDS
#  define USE_ASYNC_LDS 0
#endif

#if USE_ASYNC_LDS
// signature (from hipcc diagnostic): (v4i __device__* src, v4i __shared__* dst,
//                                     imm int offset, imm int cpol)
#define ASYNC_B128(gptr, lptr)                                                \
  __builtin_amdgcn_global_load_async_to_lds_b128(                             \
      (__attribute__((address_space(1))) v4i*)(gptr),                         \
      (__attribute__((address_space(3))) v4i*)(lptr), 0, 0)
#endif

__device__ __forceinline__ void wait_async_all() {
#if USE_ASYNC_LDS
#if __has_builtin(__builtin_amdgcn_s_wait_asynccnt)
  __builtin_amdgcn_s_wait_asynccnt(0);
#else
  asm volatile("s_wait_asynccnt 0x0" ::: "memory");
#endif
#endif
}

// ---------------------------------------------------------------- helpers ---

__device__ __forceinline__ u16 f32_to_bf16(float f) {
  unsigned int u = __float_as_uint(f);
  u += 0x7FFFu + ((u >> 16) & 1u);   // round-to-nearest-even
  return (u16)(u >> 16);
}

__device__ __forceinline__ v8f v8f_zero() {
  v8f z = {0.f, 0.f, 0.f, 0.f, 0.f, 0.f, 0.f, 0.f};
  return z;
}

// CDNA5 16-bit A/B fragment (16x32 MxK / 32x16 KxN stored "row"-contiguous):
// lane 0-15 -> row (lane&15), K chunk [k0 .. k0+7] and [k0+16 .. k0+23]
// lane 16-31 -> same rows,    K chunk [k0+8 .. k0+15] and [k0+24 .. k0+31]
__device__ __forceinline__ v16bf load_frag(const u16* base, int row0, int stride, int k0) {
  const int lane = (int)(threadIdx.x & 31u);
  const u16* p = base + (row0 + (lane & 15)) * stride + k0 + ((lane & 16) ? 8 : 0);
  union { uint4 i[2]; v16bf v; } u;
  u.i[0] = *(const uint4*)(p);
  u.i[1] = *(const uint4*)(p + 16);
  return u.v;
}

__device__ __forceinline__ v8f wmma_bf16(v16bf a, v16bf b, v8f c) {
  return __builtin_amdgcn_wmma_f32_16x16x32_bf16(false, a, false, b, (short)0, c,
                                                 false, false);
}

// ------------------------------------------------------------ f32 -> bf16 ---

__launch_bounds__(256)
__global__ void cvt_kernel(const float* __restrict__ src, u16* __restrict__ dst, int n4) {
  int i = blockIdx.x * 256 + threadIdx.x;
  if (i < n4) {
    float4 v = ((const float4*)src)[i];
    ushort4 o;
    o.x = f32_to_bf16(v.x); o.y = f32_to_bf16(v.y);
    o.z = f32_to_bf16(v.z); o.w = f32_to_bf16(v.w);
    ((ushort4*)dst)[i] = o;
  }
}

// ---------------------------------------------------------------- LayerNorm -

__launch_bounds__(256)
__global__ void ln_kernel(const float* __restrict__ x, const float* __restrict__ g,
                          const float* __restrict__ bb, u16* __restrict__ out) {
  __shared__ float r1[256];
  __shared__ float r2[256];
  const int row = blockIdx.x;
  const int tid = threadIdx.x;
  const float* xr = x + (size_t)row * DIM;
  float xv[8];
  float s1 = 0.f, s2 = 0.f;
#pragma unroll
  for (int j = 0; j < 8; ++j) {
    float v = xr[j * 256 + tid];
    xv[j] = v;
    s1 += v; s2 += v * v;
  }
  r1[tid] = s1; r2[tid] = s2;
  __syncthreads();
  for (int st = 128; st > 0; st >>= 1) {
    if (tid < st) { r1[tid] += r1[tid + st]; r2[tid] += r2[tid + st]; }
    __syncthreads();
  }
  float mu  = r1[0] * (1.0f / DIM);
  float var = r2[0] * (1.0f / DIM) - mu * mu;
  float inv = rsqrtf(var + 1e-5f);
  u16* orow = out + (size_t)row * DIM;
#pragma unroll
  for (int j = 0; j < 8; ++j) {
    int c = j * 256 + tid;
    orow[c] = f32_to_bf16((xv[j] - mu) * inv * g[c] + bb[c]);
  }
}

// ------------------------------------------------------------- WMMA GEMM ----
// C[M,N] = A[M,K](bf16,row-major) * B[K,N](bf16,row-major) + bias
// Block tile 128x256, 8 waves each 64x64 (4x4 WMMA tiles), BK=32,
// double-buffered LDS software pipeline (async global->LDS for A tiles).
// EPI: 0 = store bf16; 1 = f32 residual (out = res + C); 2 = exact GELU, bf16

template <int EPI>
__launch_bounds__(256)
__global__ void gemm_bf16_kernel(const u16* __restrict__ A, const u16* __restrict__ Bw,
                                 const float* __restrict__ bias,
                                 const float* __restrict__ res,
                                 float* __restrict__ outF, u16* __restrict__ outB,
                                 int M, int N, int K) {
  constexpr int LDT = 40;                    // 32 + 8 pad: conflict-free b128 frags
  __shared__ u16 As[2][128 * LDT];           // [row][k]
  __shared__ u16 Bs[2][256 * LDT];           // transposed: [n][k]
  const int tid  = threadIdx.x;
  const int lane = tid & 31;
  const int wave = tid >> 5;
  const int bm = blockIdx.y * 128;
  const int bn = blockIdx.x * 256;
  const int wrow = (wave & 1) * 64;          // 2 waves along M
  const int wcol = (wave >> 1) * 64;         // 4 waves along N

  const int arow = tid >> 2;                 // A: 2 chunks, rows 0..63 / 64..127
  const int acol = (tid & 3) * 8;
  const int bkr  = tid >> 5;                 // B: 4 chunks, k rows 0..7 (+8i)
  const int bcol = (tid & 31) * 8;

  v8f acc[4][4];
#pragma unroll
  for (int i = 0; i < 4; ++i)
#pragma unroll
    for (int j = 0; j < 4; ++j) acc[i][j] = v8f_zero();

  // ---- prologue: stage k=0 into buffer 0
  {
#if USE_ASYNC_LDS
#pragma unroll
    for (int i = 0; i < 2; ++i)
      ASYNC_B128(A + (size_t)(bm + arow + i * 64) * K + acol,
                 &As[0][(arow + i * 64) * LDT + acol]);
#else
#pragma unroll
    for (int i = 0; i < 2; ++i) {
      uint4 d = *(const uint4*)(A + (size_t)(bm + arow + i * 64) * K + acol);
      *(uint4*)(&As[0][(arow + i * 64) * LDT + acol]) = d;
    }
#endif
#pragma unroll
    for (int i = 0; i < 4; ++i) {
      union { uint4 v; u16 s[8]; } d;
      d.v = *(const uint4*)(Bw + (size_t)(bkr + i * 8) * N + bn + bcol);
#pragma unroll
      for (int j = 0; j < 8; ++j) Bs[0][(bcol + j) * LDT + bkr + i * 8] = d.s[j];
    }
  }

  for (int k0 = 0; k0 < K; k0 += 32) {
    const int  cur     = (k0 >> 5) & 1;
    const int  nxt     = cur ^ 1;
    const bool hasNext = (k0 + 32) < K;
    wait_async_all();
    __syncthreads();

    // stage k0+32 (global issue now; LDS scatter after compute)
    uint4 bstage[4];
#if !USE_ASYNC_LDS
    uint4 astage[2];
#endif
    if (hasNext) {
      const int kn = k0 + 32;
#if USE_ASYNC_LDS
#pragma unroll
      for (int i = 0; i < 2; ++i)
        ASYNC_B128(A + (size_t)(bm + arow + i * 64) * K + kn + acol,
                   &As[nxt][(arow + i * 64) * LDT + acol]);
#else
#pragma unroll
      for (int i = 0; i < 2; ++i)
        astage[i] = *(const uint4*)(A + (size_t)(bm + arow + i * 64) * K + kn + acol);
#endif
#pragma unroll
      for (int i = 0; i < 4; ++i)
        bstage[i] = *(const uint4*)(Bw + (size_t)(kn + bkr + i * 8) * N + bn + bcol);
    }

    // compute on cur: 16 WMMAs, each B frag feeds 4 independent WMMAs
    v16bf af[4];
#pragma unroll
    for (int mt = 0; mt < 4; ++mt) af[mt] = load_frag(&As[cur][0], wrow + mt * 16, LDT, 0);
#pragma unroll
    for (int nt = 0; nt < 4; ++nt) {
      v16bf bfr = load_frag(&Bs[cur][0], wcol + nt * 16, LDT, 0);
#pragma unroll
      for (int mt = 0; mt < 4; ++mt)
        acc[mt][nt] = wmma_bf16(af[mt], bfr, acc[mt][nt]);
    }

    if (hasNext) {
#if !USE_ASYNC_LDS
#pragma unroll
      for (int i = 0; i < 2; ++i)
        *(uint4*)(&As[nxt][(arow + i * 64) * LDT + acol]) = astage[i];
#endif
      union { uint4 v; u16 s[8]; } d;
#pragma unroll
      for (int i = 0; i < 4; ++i) {
        d.v = bstage[i];
#pragma unroll
        for (int j = 0; j < 8; ++j) Bs[nxt][(bcol + j) * LDT + bkr + i * 8] = d.s[j];
      }
    }
  }

  const int rsel = (lane & 16) ? 8 : 0;
  const int cn   = lane & 15;
#pragma unroll
  for (int mt = 0; mt < 4; ++mt) {
#pragma unroll
    for (int nt = 0; nt < 4; ++nt) {
      int   gc = bn + wcol + nt * 16 + cn;
      float bv = bias[gc];
#pragma unroll
      for (int r = 0; r < 8; ++r) {
        int    gr  = bm + wrow + mt * 16 + r + rsel;
        float  v   = acc[mt][nt][r] + bv;
        size_t idx = (size_t)gr * N + gc;
        if (EPI == 0) {
          outB[idx] = f32_to_bf16(v);
        } else if (EPI == 1) {
          outF[idx] = res[idx] + v;
        } else {
          float t = 0.5f * v * (1.0f + erff(v * 0.70710678118654752f));
          outB[idx] = f32_to_bf16(t);
        }
      }
    }
  }
}

// --------------------------------------------------------- flash attention --
// qkv: [B*S, 3*DIM] bf16 (bias already added). out: [B*S, DIM] bf16.

__launch_bounds__(256)
__global__ void attn_kernel(const u16* __restrict__ qkv, u16* __restrict__ out) {
  constexpr int LQ = 136;                    // d stride (128+8)
  constexpr int LK = 72;                     // k stride (64+8)
  __shared__ u16 Qs[128 * LQ];               // [q][d]
  __shared__ u16 Ks[64 * LQ];                // [k][d]  -> B-frag of K^T
  __shared__ u16 Vt[128 * LK];               // [d][k]  -> B-frag of V
  __shared__ u16 Ps[8 * 16 * LK];            // per-wave P [q][k]

  const int tid  = threadIdx.x;
  const int lane = tid & 31;
  const int wave = tid >> 5;
  const int qt = blockIdx.x, h = blockIdx.y, b = blockIdx.z;
  const int qbase = qt * 128;
  const size_t rowstride = 3 * DIM;
  const size_t base = (size_t)b * SEQ * rowstride;
  const float scale = 0.08838834764831845f;  // 1/sqrt(128)

  // Q tile 128x128
#pragma unroll
  for (int i = 0; i < 8; ++i) {
    int row = (tid >> 4) + i * 16;
    int d   = (tid & 15) * 8;
    const u16* g = qkv + base + (size_t)(qbase + row) * rowstride + h * HD + d;
#if USE_ASYNC_LDS
    ASYNC_B128(g, &Qs[row * LQ + d]);
#else
    *(uint4*)(&Qs[row * LQ + d]) = *(const uint4*)(g);
#endif
  }
  wait_async_all();
  __syncthreads();

  v16bf aq[4];
#pragma unroll
  for (int d0 = 0; d0 < 4; ++d0) aq[d0] = load_frag(Qs, wave * 16, LQ, d0 * 32);

  v8f o[8];
#pragma unroll
  for (int dt = 0; dt < 8; ++dt) o[dt] = v8f_zero();
  float m[8], l[8];
#pragma unroll
  for (int r = 0; r < 8; ++r) { m[r] = -1e30f; l[r] = 0.f; }

  const int rsel = (lane & 16) ? 8 : 0;
  const int cn   = lane & 15;
  u16* Pw = &Ps[wave * 16 * LK];

  const int nkt = qt * 2 + 2;                // causal: key tiles of 64 up to qbase+128
  for (int kt = 0; kt < nkt; ++kt) {
    const int kbase = kt * 64;
    __syncthreads();                         // prior tile's frag reads done
#pragma unroll
    for (int i = 0; i < 4; ++i) {
      int row = (tid >> 4) + i * 16;
      int d   = (tid & 15) * 8;
      const u16* src = qkv + base + (size_t)(kbase + row) * rowstride + h * HD + d;
#if USE_ASYNC_LDS
      ASYNC_B128(src + DIM, &Ks[row * LQ + d]);          // K slice (no transpose)
#else
      *(uint4*)(&Ks[row * LQ + d]) = *(const uint4*)(src + DIM);
#endif
      union { uint4 v; u16 s[8]; } vv;
      vv.v = *(const uint4*)(src + 2 * DIM);             // V slice -> [d][k]
#pragma unroll
      for (int j = 0; j < 8; ++j) Vt[(d + j) * LK + row] = vv.s[j];
    }
    wait_async_all();
    __syncthreads();

    // S = Q K^T (16x64 per wave)
    v8f s[4];
#pragma unroll
    for (int nt = 0; nt < 4; ++nt) {
      s[nt] = v8f_zero();
#pragma unroll
      for (int d0 = 0; d0 < 4; ++d0) {
        v16bf bk = load_frag(Ks, nt * 16, LQ, d0 * 32);
        s[nt] = wmma_bf16(aq[d0], bk, s[nt]);
      }
    }

    // online softmax per row slot
#pragma unroll
    for (int r = 0; r < 8; ++r) {
      const int q = qbase + wave * 16 + r + rsel;
      float sm = -1e30f;
      float sv[4];
#pragma unroll
      for (int nt = 0; nt < 4; ++nt) {
        int   kc = kbase + nt * 16 + cn;
        float xx = s[nt][r] * scale;
        xx = (kc <= q) ? xx : -1e30f;
        sv[nt] = xx;
        sm = fmaxf(sm, xx);
      }
#pragma unroll
      for (int off = 1; off < 16; off <<= 1) sm = fmaxf(sm, __shfl_xor(sm, off, 32));
      float mn    = fmaxf(m[r], sm);
      float alpha = __expf(m[r] - mn);
      m[r] = mn;
      float rs = 0.f;
#pragma unroll
      for (int nt = 0; nt < 4; ++nt) {
        int   kc = kbase + nt * 16 + cn;
        float p  = (kc <= q) ? __expf(sv[nt] - mn) : 0.f;   // exact 0 when masked
        rs += p;
        Pw[(r + rsel) * LK + nt * 16 + cn] = f32_to_bf16(p);
      }
#pragma unroll
      for (int off = 1; off < 16; off <<= 1) rs += __shfl_xor(rs, off, 32);
      l[r] = l[r] * alpha + rs;
#pragma unroll
      for (int dt = 0; dt < 8; ++dt) o[dt][r] *= alpha;
    }

    // O += P @ V
    v16bf ap0 = load_frag(Pw, 0, LK, 0);
    v16bf ap1 = load_frag(Pw, 0, LK, 32);
#pragma unroll
    for (int dt = 0; dt < 8; ++dt) {
      v16bf bv0 = load_frag(Vt, dt * 16, LK, 0);
      v16bf bv1 = load_frag(Vt, dt * 16, LK, 32);
      o[dt] = wmma_bf16(ap0, bv0, o[dt]);
      o[dt] = wmma_bf16(ap1, bv1, o[dt]);
    }
  }

  // normalize + write [B,S,NH,HD] flattened
#pragma unroll
  for (int r = 0; r < 8; ++r) {
    float  inv  = 1.0f / l[r];
    int    q    = qbase + wave * 16 + r + rsel;
    size_t orow = ((size_t)b * SEQ + q) * DIM + h * HD;
#pragma unroll
    for (int dt = 0; dt < 8; ++dt)
      out[orow + dt * 16 + cn] = f32_to_bf16(o[dt][r] * inv);
  }
}

// ---------------------------------------------------------------- launcher --

extern "C" void kernel_launch(void* const* d_in, const int* in_sizes, int n_in,
                              void* d_out, int out_size, void* d_ws, size_t ws_size,
                              hipStream_t stream) {
  (void)in_sizes; (void)n_in; (void)out_size; (void)ws_size;
  const float* x      = (const float*)d_in[0];
  // d_in[1] = attention_mask (causal; implemented analytically)
  const float* ln1_g  = (const float*)d_in[2];
  const float* ln1_b  = (const float*)d_in[3];
  const float* attn_w = (const float*)d_in[4];
  const float* attn_b = (const float*)d_in[5];
  const float* proj_w = (const float*)d_in[6];
  const float* proj_b = (const float*)d_in[7];
  const float* ln2_g  = (const float*)d_in[8];
  const float* ln2_b  = (const float*)d_in[9];
  const float* fc_w   = (const float*)d_in[10];
  const float* fc_b   = (const float*)d_in[11];
  const float* out_w  = (const float*)d_in[12];
  const float* out_b  = (const float*)d_in[13];
  float* y = (float*)d_out;

  // workspace carve-out (~305 MB total)
  char* p = (char*)d_ws;
  auto alloc = [&](size_t bytes) {
    char* q = p;
    p += (bytes + 255) & ~(size_t)255;
    return q;
  };
  u16*  wq  = (u16*)alloc((size_t)2048 * 6144 * 2);
  u16*  wp  = (u16*)alloc((size_t)2048 * 2048 * 2);
  u16*  wf  = (u16*)alloc((size_t)2048 * 8192 * 2);
  u16*  wo  = (u16*)alloc((size_t)8192 * 2048 * 2);
  u16*  h1  = (u16*)alloc((size_t)TOKENS * 2048 * 2);
  u16*  qkv = (u16*)alloc((size_t)TOKENS * 6144 * 2);
  u16*  att = (u16*)alloc((size_t)TOKENS * 2048 * 2);
  float* x1 = (float*)alloc((size_t)TOKENS * 2048 * 4);
  u16*  h2  = (u16*)alloc((size_t)TOKENS * 2048 * 2);
  u16*  fca = (u16*)alloc((size_t)TOKENS * 8192 * 2);

  // weights -> bf16 (every call: launcher must be stateless/deterministic)
  cvt_kernel<<<2048 * 6144 / 1024, 256, 0, stream>>>(attn_w, wq, 2048 * 6144 / 4);
  cvt_kernel<<<2048 * 2048 / 1024, 256, 0, stream>>>(proj_w, wp, 2048 * 2048 / 4);
  cvt_kernel<<<2048 * 8192 / 1024, 256, 0, stream>>>(fc_w,   wf, 2048 * 8192 / 4);
  cvt_kernel<<<8192 * 2048 / 1024, 256, 0, stream>>>(out_w,  wo, 8192 * 2048 / 4);

  // h1 = LN1(x)
  ln_kernel<<<TOKENS, 256, 0, stream>>>(x, ln1_g, ln1_b, h1);
  // qkv = h1 @ attn_w + attn_b
  gemm_bf16_kernel<0><<<dim3(6144 / 256, TOKENS / 128), 256, 0, stream>>>(
      h1, wq, attn_b, nullptr, nullptr, qkv, TOKENS, 6144, 2048);
  // att = softmax(QK^T/sqrt(d) + causal) V
  attn_kernel<<<dim3(SEQ / 128, NH, BATCH), 256, 0, stream>>>(qkv, att);
  // x1 = x + att @ proj_w + proj_b
  gemm_bf16_kernel<1><<<dim3(2048 / 256, TOKENS / 128), 256, 0, stream>>>(
      att, wp, proj_b, x, x1, nullptr, TOKENS, 2048, 2048);
  // h2 = LN2(x1)
  ln_kernel<<<TOKENS, 256, 0, stream>>>(x1, ln2_g, ln2_b, h2);
  // fca = gelu(h2 @ fc_w + fc_b)
  gemm_bf16_kernel<2><<<dim3(8192 / 256, TOKENS / 128), 256, 0, stream>>>(
      h2, wf, fc_b, nullptr, nullptr, fca, TOKENS, 8192, 2048);
  // y = x1 + fca @ out_w + out_b
  gemm_bf16_kernel<1><<<dim3(2048 / 256, TOKENS / 128), 256, 0, stream>>>(
      fca, wo, out_b, x1, y, nullptr, TOKENS, 2048, 8192);
}
